// BinaryConv_39359080300786
// MI455X (gfx1250) — matrix operand order repaired and checked by reading, not compiled, not run on hardware
//
#include <hip/hip_runtime.h>
#include <hip/hip_bf16.h>

// ---------------------------------------------------------------------------
// BinaryConv on MI455X (gfx1250): implicit-GEMM with v_wmma_f32_16x16x32_f16
// + global_load_async_to_lds_b128 for the weight-sign tile.
//
// out[n,o,h,w] = alpha[o] * sum_{c,kh,kw} sign(w[o,c,kh,kw]) * x[n,c,h+kh-1,w+kw-1]
//
// GEMM view: M = O = 256, P = 32*56*56 = 100352, K = 256*9 = 2304.
// K-loop split as (tap 0..8) x (c-chunk of 32). Weights are exact +/-1 in f16
// (products exact), f32 accumulate, alpha applied in f32 at the epilogue.
// ---------------------------------------------------------------------------

typedef _Float16 v16h __attribute__((ext_vector_type(16)));
typedef _Float16 v8h  __attribute__((ext_vector_type(8)));
typedef float    v8f  __attribute__((ext_vector_type(8)));

#define NIMG   32
#define CIN    256
#define COUT   256
#define HH     56
#define WW     56
#define HWSZ   (HH * WW)          // 3136
#define PTOT   (NIMG * HWSZ)      // 100352
#define KTAPS  9

#define BM     128                // output-channel tile
#define BP     128                // spatial tile
#define BK     32                 // K chunk (channels per wmma group)
#define LDSP   40                 // LDS row pitch in halves -> 80B rows, 16B aligned

// LDS pool layout (single pool => known group-segment byte offsets for async dst)
#define AT_OFF     0
#define AT_BYTES   (BM * LDSP * 2)            // 10240
#define BT_OFF     AT_BYTES                   // 10240
#define BT_BYTES   (BP * LDSP * 2)            // 10240
#define ALPHA_OFF  (BT_OFF + BT_BYTES)        // 20480
#define SMEM_BYTES (ALPHA_OFF + BM * 4)       // 20992

// ---------------------------------------------------------------------------
// Phase 1: alpha[o] = mean(|w[o,:,:,:]|); signs as f16 in Wr[tap][o][c].
// ---------------------------------------------------------------------------
__global__ __launch_bounds__(256)
void bconv_binarize_kernel(const float* __restrict__ w,
                           float* __restrict__ alpha,
                           _Float16* __restrict__ Wr) {
    __shared__ float red[256];
    const int o = blockIdx.x;
    const int t = threadIdx.x;
    const float* wo = w + (size_t)o * (CIN * KTAPS);   // 2304 elements

    float s = 0.0f;
    float vals[9];
#pragma unroll
    for (int i = 0; i < 9; ++i) {
        const int k = t + i * 256;
        const float v = wo[k];
        vals[i] = v;
        s += fabsf(v);
    }
    red[t] = s;
    __syncthreads();
#pragma unroll
    for (int stride = 128; stride > 0; stride >>= 1) {
        if (t < stride) red[t] += red[t + stride];
        __syncthreads();
    }
    if (t == 0) alpha[o] = red[0] * (1.0f / (CIN * KTAPS));

#pragma unroll
    for (int i = 0; i < 9; ++i) {
        const int k   = t + i * 256;
        const int c   = k / KTAPS;
        const int tap = k - c * KTAPS;
        const float v = vals[i];
        const float sgn = (v > 0.0f) ? 1.0f : ((v < 0.0f) ? -1.0f : 0.0f);
        Wr[((size_t)tap * COUT + o) * CIN + c] = (_Float16)sgn;
    }
}

// ---------------------------------------------------------------------------
// Phase 2: implicit-GEMM conv. Grid: (PTOT/BP, COUT/BM). 256 threads = 8 waves,
// waves 0-3 stage B (im2col f32->f16), waves 4-7 stage A via async-to-LDS.
// All 8 waves (4Mx2P grid) run 2x4 = 8 WMMAs per 32-deep K chunk.
// ---------------------------------------------------------------------------
__global__ __launch_bounds__(256)
void bconv_gemm_kernel(const float* __restrict__ x,
                       const _Float16* __restrict__ Wr,
                       const float* __restrict__ alpha,
                       float* __restrict__ out) {
    __shared__ alignas(16) unsigned char smem[SMEM_BYTES];
    _Float16* At     = (_Float16*)(smem + AT_OFF);     // [BM][LDSP]
    _Float16* Bt     = (_Float16*)(smem + BT_OFF);     // [BP][LDSP]
    float*    alphaS = (float*)(smem + ALPHA_OFF);     // [BM]

    const int tid   = threadIdx.x;
    const int Pbase = blockIdx.x * BP;
    const int Mbase = blockIdx.y * BM;

    const int waveId = tid >> 5;
    const int lane   = tid & 31;
    const int wm     = waveId & 3;       // 0..3 -> M offset wm*32
    const int wn     = waveId >> 2;      // 0..1 -> P offset wn*64
    const int mrow   = lane & 15;
    const int hiH    = (lane >> 4) & 1;  // lane 16..31

    if (tid < BM) alphaS[tid] = alpha[Mbase + tid];

    // -------- per-thread loop-invariant im2col state (threads 0..127) -------
    // thread tid owns spatial position p = Pbase + tid and Bt row tid.
    int ph = 0, pw = 0;
    const float* xpix = nullptr;         // &x[ni][0][0][0]
    _Float16*    btRow = nullptr;
    if (tid < BP) {
        const int p  = Pbase + tid;
        const int ni = p / HWSZ;
        const int hw = p - ni * HWSZ;
        ph = hw / WW;
        pw = hw - ph * WW;
        xpix  = x + (size_t)ni * CIN * HWSZ;
        btRow = Bt + tid * LDSP;
    }

    v8f acc[2][4];
#pragma unroll
    for (int mt = 0; mt < 2; ++mt)
#pragma unroll
        for (int nt = 0; nt < 4; ++nt)
            acc[mt][nt] = (v8f)(0.0f);

    for (int tap = 0; tap < KTAPS; ++tap) {
        const int kh = tap / 3;
        const int kw = tap - kh * 3;

        // per-tap shifted pixel (threads 0..127)
        bool valid = false;
        const float* xrow = nullptr;
        if (tid < BP) {
            const int hi = ph + kh - 1;
            const int wi = pw + kw - 1;
            valid = ((unsigned)hi < (unsigned)HH) && ((unsigned)wi < (unsigned)WW);
            xrow  = xpix + hi * WW + wi;     // + c*HWSZ selects the channel
        }
        const _Float16* Wtap = Wr + (((size_t)tap * COUT + Mbase) * CIN);

        for (int kc = 0; kc < CIN; kc += BK) {
            __syncthreads();   // previous tiles fully consumed

            if (tid < BP) {
                // ---- B tile row: 32 channels, immediate-offset f32 loads ----
                const float* xc = xrow + (size_t)kc * HWSZ;
                if (valid) {
#pragma unroll
                    for (int cl = 0; cl < BK; cl += 8) {
                        v8h hv;
#pragma unroll
                        for (int j = 0; j < 8; ++j)
                            hv[j] = (_Float16)xc[(size_t)(cl + j) * HWSZ];
                        *(v8h*)(btRow + cl) = hv;
                    }
                } else {
                    const v8h z = (v8h)((_Float16)0.0f);
#pragma unroll
                    for (int cl = 0; cl < BK; cl += 8)
                        *(v8h*)(btRow + cl) = z;
                }
            } else {
                // ---- A tile: 128 rows x 64B via async global->LDS DMA ----
                const int t2 = tid - BP;                 // 0..127
                const _Float16* Wt = Wtap + kc;
#pragma unroll
                for (int s = 0; s < 4; ++s) {
                    const int seg = t2 + s * 128;        // 0..511
                    const int row = seg >> 2;            // 0..127
                    const int q   = seg & 3;             // 16B segment in row
                    const unsigned long long gaddr =
                        (unsigned long long)(const void*)(Wt + (size_t)row * CIN + q * 8);
                    const unsigned ldsoff = AT_OFF + row * (LDSP * 2) + q * 16;
                    asm volatile("global_load_async_to_lds_b128 %0, %1, off"
                                 :: "v"(ldsoff), "v"(gaddr)
                                 : "memory");
                }
            }
            asm volatile("s_wait_asynccnt 0x0" ::: "memory");
            __syncthreads();

            // ---- fragments per documented CDNA5 wave32 layouts ----
            // A (16x32 f16): lane<16 -> M=lane, K 0-7 & 16-23; lane>=16 -> K 8-15 & 24-31
            v16h afr[2];
#pragma unroll
            for (int mt = 0; mt < 2; ++mt) {
                const int r  = wm * 32 + mt * 16 + mrow;
                const int kb = hiH ? 8 : 0;
                const v8h a0 = *(const v8h*)(&At[r * LDSP + kb]);
                const v8h a1 = *(const v8h*)(&At[r * LDSP + kb + 16]);
                afr[mt] = __builtin_shufflevector(a0, a1,
                          0,1,2,3,4,5,6,7,8,9,10,11,12,13,14,15);
            }
            // B (32x16 f16): lane<16 -> N=lane, K 0-7 & 16-23; lane>=16 -> K 8-15 & 24-31
            v16h bfr[4];
#pragma unroll
            for (int nt = 0; nt < 4; ++nt) {
                const int pr = wn * 64 + nt * 16 + mrow;
                const int kb = hiH ? 8 : 0;
                const v8h b0 = *(const v8h*)(&Bt[pr * LDSP + kb]);
                const v8h b1 = *(const v8h*)(&Bt[pr * LDSP + kb + 16]);
                bfr[nt] = __builtin_shufflevector(b0, b1,
                          0,1,2,3,4,5,6,7,8,9,10,11,12,13,14,15);
            }

#pragma unroll
            for (int mt = 0; mt < 2; ++mt)
#pragma unroll
                for (int nt = 0; nt < 4; ++nt)
                    acc[mt][nt] = __builtin_amdgcn_wmma_f32_16x16x32_f16(
                        false, afr[mt], false, bfr[nt],
                        (short)0, acc[mt][nt], false, false);
        }
    }
    __syncthreads();

    // ---- epilogue: scale by alpha[o], scatter to NCHW ----
    // C/D layout: lane<16 -> col=lane, M=v; lane>=16 -> col=lane-16, M=v+8
#pragma unroll
    for (int nt = 0; nt < 4; ++nt) {
        const int p  = Pbase + wn * 64 + nt * 16 + mrow;
        const int ni = p / HWSZ;
        const int hw = p - ni * HWSZ;
        float* outp = out + ((size_t)ni * COUT) * HWSZ + hw;
#pragma unroll
        for (int mt = 0; mt < 2; ++mt) {
            const int mloc = wm * 32 + mt * 16 + hiH * 8;
#pragma unroll
            for (int v = 0; v < 8; ++v) {
                const int oLoc = mloc + v;                 // 0..127 in tile
                const int o    = Mbase + oLoc;
                outp[(size_t)o * HWSZ] = acc[mt][nt][v] * alphaS[oLoc];
            }
        }
    }
}

// ---------------------------------------------------------------------------
extern "C" void kernel_launch(void* const* d_in, const int* in_sizes, int n_in,
                              void* d_out, int out_size, void* d_ws, size_t ws_size,
                              hipStream_t stream) {
    const float* x = (const float*)d_in[0];   // [32,256,56,56]
    const float* w = (const float*)d_in[1];   // [256,256,3,3]
    float* out = (float*)d_out;               // [32,256,56,56]

    float*    alpha = (float*)d_ws;                            // 256 f32
    _Float16* Wr    = (_Float16*)((char*)d_ws + 1024);         // [9][256][256] f16

    bconv_binarize_kernel<<<COUT, 256, 0, stream>>>(w, alpha, Wr);

    dim3 grid(PTOT / BP, COUT / BM);          // (784, 2)
    bconv_gemm_kernel<<<grid, 256, 0, stream>>>(x, Wr, alpha, out);
}
